// SparseMoEBlock_1726576854834
// MI455X (gfx1250) — matrix operand order repaired and checked by reading, not compiled, not run on hardware
//
#include <hip/hip_runtime.h>
#include <hip/hip_bf16.h>

// SparseMoE for MI455X (gfx1250): bf16 WMMA routed grouped-GEMM,
// full-FF fusion per 32-token tile using CDNA5's 320KB-per-WGP LDS.
// T=8192 tokens, D=1024, FF=3584, E=8, top-2.
//
// Roofline: routed FLOPs ~353 GFLOP (bf16 WMMA-bound); bf16 weights (176 MB)
// fit in the 192 MB global L2, so the 11 GB of per-expert weight re-reads are
// L2 traffic, not HBM. Full-FF fusion keeps h[32,3584] in LDS and reduces the
// down-projection locally, so output atomics are only the top-2 merge (64 MB).

#define T_   8192
#define D_   1024
#define FF_  3584
#define E_   8

#define A_STRIDE 1032          // halves; 2064 B row stride (16B multiple)
#define H_STRIDE 3600          // halves; 7200 B row stride (16B multiple)
#define SMEM_BYTES ((32 * A_STRIDE + 32 * H_STRIDE) * 2 + 32 * 4 + 32 * 4)  // 296,704 B

typedef __attribute__((ext_vector_type(16))) __bf16        v16bf;
typedef __attribute__((ext_vector_type(8)))  float         v8f;
typedef __attribute__((ext_vector_type(4)))  unsigned int  u32x4;

union Frag16 { u32x4 q[2]; v16bf v; };   // 32 bytes = 16 bf16 = one WMMA A/B fragment

__device__ __forceinline__ unsigned short f2bf(float f) {
  unsigned u = __float_as_uint(f);
  u += 0x7FFFu + ((u >> 16) & 1u);       // round-to-nearest-even
  return (unsigned short)(u >> 16);
}

// ---------------------------------------------------------------- f32 -> bf16
__global__ void cvt_f32_bf16_x4(const float* __restrict__ src,
                                unsigned short* __restrict__ dst, int n4) {
  int i = blockIdx.x * blockDim.x + threadIdx.x;
  if (i >= n4) return;
  float4 v = ((const float4*)src)[i];
  ushort4 o;
  o.x = f2bf(v.x); o.y = f2bf(v.y); o.z = f2bf(v.z); o.w = f2bf(v.w);
  ((ushort4*)dst)[i] = o;
}

// ---------------------------------------------------------------- router
__global__ __launch_bounds__(256) void router_kernel(
    const float* __restrict__ x, const float* __restrict__ rw,
    float* __restrict__ logits, int* __restrict__ cnt,
    int* __restrict__ idxl, float* __restrict__ gwl) {
  __shared__ float rw_s[E_ * D_];
  for (int i = threadIdx.x; i < E_ * D_; i += 256) rw_s[i] = rw[i];
  __syncthreads();

  int t = blockIdx.x * 256 + threadIdx.x;
  float acc[E_];
#pragma unroll
  for (int e = 0; e < E_; ++e) acc[e] = 0.f;

  const float4* xr = (const float4*)(x + (size_t)t * D_);
  for (int d4 = 0; d4 < D_ / 4; ++d4) {
    float4 xv = xr[d4];
#pragma unroll
    for (int e = 0; e < E_; ++e) {
      float4 wv = ((const float4*)(rw_s + e * D_))[d4];
      acc[e] += xv.x * wv.x + xv.y * wv.y + xv.z * wv.z + xv.w * wv.w;
    }
  }
#pragma unroll
  for (int e = 0; e < E_; ++e) logits[(size_t)t * E_ + e] = acc[e];

  float mx = acc[0];
#pragma unroll
  for (int e = 1; e < E_; ++e) mx = fmaxf(mx, acc[e]);
  float p[E_], s = 0.f;
#pragma unroll
  for (int e = 0; e < E_; ++e) { p[e] = __expf(acc[e] - mx); s += p[e]; }
  float inv = 1.f / s;
#pragma unroll
  for (int e = 0; e < E_; ++e) p[e] *= inv;

  int i1 = 0; float v1 = p[0];
#pragma unroll
  for (int e = 1; e < E_; ++e) if (p[e] > v1) { v1 = p[e]; i1 = e; }
  int i2 = 0; float v2 = -1.f;
#pragma unroll
  for (int e = 0; e < E_; ++e) if (e != i1 && p[e] > v2) { v2 = p[e]; i2 = e; }

  int s1 = atomicAdd(&cnt[i1], 1);
  idxl[i1 * T_ + s1] = t; gwl[i1 * T_ + s1] = v1;
  int s2 = atomicAdd(&cnt[i2], 1);
  idxl[i2 * T_ + s2] = t; gwl[i2 * T_ + s2] = v2;
}

// ---------------------------------------------------------------- WMMA helper
__device__ __forceinline__ v8f wmma_bf16(v16bf a, v16bf b, v8f c) {
  return __builtin_amdgcn_wmma_f32_16x16x32_bf16(
      false, a, false, b, (short)0, c, false, false);
}

// ---------------------------------------------------------------- fused expert
// grid = (T/32 tile slots, E), block = 256 (8 waves), ~290KB dynamic LDS.
// Stage 0: gather 32 bf16 token rows (A tile) into LDS.
// Phase 1: h[32,3584] = silu(A@wg^T)*(A@wu^T)*gate_w, kept entirely in LDS.
// Phase 2: out[32,1024] = h @ wd^T reduced over full FF locally; only the
//          per-token top-2 expert merge needs f32 atomics.
__global__ __launch_bounds__(256) void moe_expert_kernel(
    const unsigned short* __restrict__ xb,  const unsigned short* __restrict__ wgb,
    const unsigned short* __restrict__ wub, const unsigned short* __restrict__ wdb,
    const int* __restrict__ cnt, const int* __restrict__ idxl,
    const float* __restrict__ gwl, float* __restrict__ out) {
  const int e    = blockIdx.y;
  const int tile = blockIdx.x;
  const int n_e  = cnt[e];
  if (tile * 32 >= n_e) return;        // uniform early-exit, before any barrier

  extern __shared__ __align__(16) unsigned short smem[];
  unsigned short* a_lds = smem;                         // 32 x A_STRIDE
  unsigned short* h_lds = smem + 32 * A_STRIDE;         // 32 x H_STRIDE
  int*   tok_s = (int*)(h_lds + 32 * H_STRIDE);         // 32
  float* gw_s  = (float*)(tok_s + 32);                  // 32

  if (threadIdx.x < 32) {
    int s = tile * 32 + threadIdx.x;
    if (s < n_e) { tok_s[threadIdx.x] = idxl[e * T_ + s]; gw_s[threadIdx.x] = gwl[e * T_ + s]; }
    else         { tok_s[threadIdx.x] = 0;               gw_s[threadIdx.x] = 0.f; }
  }
  __syncthreads();

  // ---- stage 0: gather A tile (32 x 1024 bf16) into LDS ----
  for (int c = threadIdx.x; c < 32 * (D_ / 8); c += 256) {   // 8 halves per chunk
    int row = c >> 7, col = (c & 127) << 3;
    u32x4 v = *(const u32x4*)(xb + (size_t)tok_s[row] * D_ + col);
    *(u32x4*)(a_lds + row * A_STRIDE + col) = v;
  }
  __syncthreads();

  const int lane = threadIdx.x & 31;
  const int w    = threadIdx.x >> 5;
  const int m    = lane & 15;                 // A row / B column index for this lane
  const int o    = (lane < 16) ? 0 : 8;       // A frag K sub-offset (ISA 16-bit A layout)
  const int ko   = (lane < 16) ? 0 : 8;       // B frag K sub-offset (ISA 16-bit B layout)
  const int mb   = (lane < 16) ? 0 : 8;       // C/D row base for this lane half

  const unsigned short* arow0 = a_lds + m * A_STRIDE;
  const unsigned short* arow1 = a_lds + (16 + m) * A_STRIDE;

  // ---- phase 1: gate/up GEMM over K=D, 224 col-tiles split across 8 waves ----
  const size_t wbase = (size_t)e * FF_ * D_;
  for (int ct = w; ct < FF_ / 16; ct += 8) {
    const int fbase = ct * 16;
    const unsigned short* gptr = wgb + wbase + (size_t)(fbase + m) * D_;
    const unsigned short* uptr = wub + wbase + (size_t)(fbase + m) * D_;

    v8f g0 = {}, g1 = {}, u0 = {}, u1 = {};
    for (int kb = 0; kb < D_; kb += 32) {
      Frag16 A0, A1, Bg, Bu;
      A0.q[0] = *(const u32x4*)(arow0 + kb + o);
      A0.q[1] = *(const u32x4*)(arow0 + kb + o + 16);
      A1.q[0] = *(const u32x4*)(arow1 + kb + o);
      A1.q[1] = *(const u32x4*)(arow1 + kb + o + 16);
      Bg.q[0] = *(const u32x4*)(gptr + kb + ko);
      Bg.q[1] = *(const u32x4*)(gptr + kb + ko + 16);
      Bu.q[0] = *(const u32x4*)(uptr + kb + ko);
      Bu.q[1] = *(const u32x4*)(uptr + kb + ko + 16);
      g0 = wmma_bf16(A0.v, Bg.v, g0);
      g1 = wmma_bf16(A1.v, Bg.v, g1);
      u0 = wmma_bf16(A0.v, Bu.v, u0);
      u1 = wmma_bf16(A1.v, Bu.v, u1);
    }
#pragma unroll
    for (int r = 0; r < 8; ++r) {
      float ga = g0[r], ua = u0[r];
      float gb = g1[r], ub = u1[r];
      float h0 = (ga / (1.f + __expf(-ga))) * ua * gw_s[mb + r];
      float h1 = (gb / (1.f + __expf(-gb))) * ub * gw_s[16 + mb + r];
      h_lds[(mb + r) * H_STRIDE + fbase + m]      = f2bf(h0);
      h_lds[(16 + mb + r) * H_STRIDE + fbase + m] = f2bf(h1);
    }
  }

  int tokr0[8], tokr1[8];
#pragma unroll
  for (int r = 0; r < 8; ++r) { tokr0[r] = tok_s[mb + r]; tokr1[r] = tok_s[16 + mb + r]; }
  __syncthreads();

  // ---- phase 2: out[32,1024] += h[32,3584] @ wd^T, full-FF reduction ----
  const unsigned short* hrow0 = h_lds + m * H_STRIDE;
  const unsigned short* hrow1 = h_lds + (16 + m) * H_STRIDE;
  const unsigned short* wde   = wdb + (size_t)e * D_ * FF_;

  for (int np = 0; np < 4; ++np) {          // 8 dcol tiles per wave, in pairs
    const int nt0 = w * 8 + np * 2, nt1 = nt0 + 1;
    const unsigned short* bp0 = wde + (size_t)(nt0 * 16 + m) * FF_;
    const unsigned short* bp1 = wde + (size_t)(nt1 * 16 + m) * FF_;

    v8f c00 = {}, c01 = {}, c10 = {}, c11 = {};
    for (int kb = 0; kb < FF_; kb += 32) {
      Frag16 A0, A1, B0, B1;
      A0.q[0] = *(const u32x4*)(hrow0 + kb + o);
      A0.q[1] = *(const u32x4*)(hrow0 + kb + o + 16);
      A1.q[0] = *(const u32x4*)(hrow1 + kb + o);
      A1.q[1] = *(const u32x4*)(hrow1 + kb + o + 16);
      B0.q[0] = *(const u32x4*)(bp0 + kb + ko);
      B0.q[1] = *(const u32x4*)(bp0 + kb + ko + 16);
      B1.q[0] = *(const u32x4*)(bp1 + kb + ko);
      B1.q[1] = *(const u32x4*)(bp1 + kb + ko + 16);
      c00 = wmma_bf16(A0.v, B0.v, c00);
      c10 = wmma_bf16(A1.v, B0.v, c10);
      c01 = wmma_bf16(A0.v, B1.v, c01);
      c11 = wmma_bf16(A1.v, B1.v, c11);
    }
#pragma unroll
    for (int r = 0; r < 8; ++r) {
      unsafeAtomicAdd(out + (size_t)tokr0[r] * D_ + nt0 * 16 + m, c00[r]);
      unsafeAtomicAdd(out + (size_t)tokr1[r] * D_ + nt0 * 16 + m, c10[r]);
      unsafeAtomicAdd(out + (size_t)tokr0[r] * D_ + nt1 * 16 + m, c01[r]);
      unsafeAtomicAdd(out + (size_t)tokr1[r] * D_ + nt1 * 16 + m, c11[r]);
    }
  }
}

// ---------------------------------------------------------------- launch
extern "C" void kernel_launch(void* const* d_in, const int* in_sizes, int n_in,
                              void* d_out, int out_size, void* d_ws, size_t ws_size,
                              hipStream_t stream) {
  const float* x  = (const float*)d_in[0];
  const float* rw = (const float*)d_in[1];
  const float* wg = (const float*)d_in[2];
  const float* wu = (const float*)d_in[3];
  const float* wd = (const float*)d_in[4];
  float* out    = (float*)d_out;
  float* logits = out + (size_t)T_ * D_;   // tuple: (out, router_logits) concatenated

  // workspace layout (~193 MB)
  char* ws = (char*)d_ws;
  int*   cnt  = (int*)ws;                                       // 256 B reserved
  int*   idxl = (int*)(ws + 256);                               // E*T ints
  float* gwl  = (float*)(ws + 256 + (size_t)E_ * T_ * 4);       // E*T floats
  unsigned short* xb  = (unsigned short*)(ws + 256 + (size_t)E_ * T_ * 8);
  unsigned short* wgb = xb  + (size_t)T_ * D_;
  unsigned short* wub = wgb + (size_t)E_ * FF_ * D_;
  unsigned short* wdb = wub + (size_t)E_ * FF_ * D_;

  // Host-side attribute; idempotent and identical every call (no static guards).
  hipFuncSetAttribute((const void*)moe_expert_kernel,
                      hipFuncAttributeMaxDynamicSharedMemorySize, SMEM_BYTES);

  hipMemsetAsync(cnt, 0, 256, stream);
  hipMemsetAsync(out, 0, (size_t)T_ * D_ * sizeof(float), stream);

  {
    int n4 = T_ * D_ / 4;
    cvt_f32_bf16_x4<<<(n4 + 255) / 256, 256, 0, stream>>>(x, xb, n4);
  }
  {
    int n4 = E_ * FF_ * D_ / 4;
    int nb = (n4 + 255) / 256;
    cvt_f32_bf16_x4<<<nb, 256, 0, stream>>>(wg, wgb, n4);
    cvt_f32_bf16_x4<<<nb, 256, 0, stream>>>(wu, wub, n4);
    cvt_f32_bf16_x4<<<nb, 256, 0, stream>>>(wd, wdb, n4);
  }

  router_kernel<<<T_ / 256, 256, 0, stream>>>(x, rw, logits, cnt, idxl, gwl);

  dim3 grid(T_ / 32, E_);
  moe_expert_kernel<<<grid, 256, SMEM_BYTES, stream>>>(xb, wgb, wub, wdb, cnt, idxl, gwl, out);
}
